// IntraAttention_37787122270764
// MI455X (gfx1250) — matrix operand order repaired and checked by reading, not compiled, not run on hardware
//
#include <hip/hip_runtime.h>

#define BQ 8
#define SQ 2048
#define DQ 1024
#define MTOT (BQ * SQ) // 16384

typedef __bf16 bf16_t;
typedef bf16_t v16bf __attribute__((ext_vector_type(16)));
typedef float  v8f   __attribute__((ext_vector_type(8)));
typedef unsigned int v8u __attribute__((ext_vector_type(8)));
typedef unsigned int u32x4 __attribute__((ext_vector_type(4)));
typedef int i32x4 __attribute__((ext_vector_type(4)));
typedef int i32x8 __attribute__((ext_vector_type(8)));

// ---------- helpers ----------

__device__ __forceinline__ unsigned short f2bf(float f) {
  unsigned int u = __float_as_uint(f);
  u = (u + 0x7FFFu + ((u >> 16) & 1u)) >> 16; // round-to-nearest-even
  return (unsigned short)u;
}

__device__ __forceinline__ v8f wmma_bf16(v16bf a, v16bf b, v8f c) {
  // D = A(16x32) * B(32x16) + C, f32 accumulate
  return __builtin_amdgcn_wmma_f32_16x16x32_bf16(
      false, a, false, b, (short)0, c, false, false);
}

// A fragment 16(M) x 32(K), row-major bf16 source, leading dim `ld` elems.
// base points at (row0, k0).  ISA 16-bit A layout:
//   lanes 0-15 (hi=0): v0..3 -> K{0..7}, v4..7 -> K{16..23}
//   lanes 16-31 (hi=1): +8
__device__ __forceinline__ v16bf load_a_rowmajor(const unsigned short* base,
                                                 int ld, int lane) {
  const int m = lane & 15, hi = lane >> 4;
  const unsigned short* p = base + (size_t)m * ld + 8 * hi;
  v8u r;
#pragma unroll
  for (int v = 0; v < 8; ++v) {
    const int kk = (v < 4) ? (2 * v) : (8 + 2 * v); // 2v | 16+2(v-4)
    r[v] = *(const unsigned int*)(p + kk);
  }
  return __builtin_bit_cast(v16bf, r);
}

// B fragment 32(K) x 16(N), "k-major" source: element(k,n) at base + n*ldn + k.
// ISA 16-bit B layout (same as A): lanes 0-15: v0..3 -> K{0..7}, v4..7 -> K{16..23}; lanes 16-31: +8.
__device__ __forceinline__ v16bf load_b_kmajor(const unsigned short* base,
                                               int ldn, int lane) {
  const int n = lane & 15, hi = lane >> 4;
  const unsigned short* p = base + (size_t)n * ldn + 8 * hi;
  v8u r;
#pragma unroll
  for (int v = 0; v < 8; ++v) r[v] = *(const unsigned int*)(p + ((v < 4) ? (2 * v) : (8 + 2 * v)));
  return __builtin_bit_cast(v16bf, r);
}

// TDM: DMA a 128-row x 32-col bf16 tile (row stride 1024 elems) into LDS.
// D# per CDNA5 ISA ch.8: group0 = {flags, lds_addr, global_addr, type=2},
// group1 = {data_size=2B, tensor dims, tile dims 32x128, dim0 stride}.
__device__ __forceinline__ void tdm_load_tile_a(unsigned short* ldsdst,
                                                const unsigned short* gsrc) {
  unsigned long long ga = (unsigned long long)(size_t)gsrc;
  unsigned int lds = (unsigned int)(size_t)ldsdst;
  u32x4 g0;
  g0[0] = 1u;                       // count=1, user descriptor, no gather
  g0[1] = lds;                      // lds_addr (bytes)
  g0[2] = (unsigned int)ga;         // global_addr[31:0]
  g0[3] = (unsigned int)((ga >> 32) & 0x01FFFFFFull) | 0x80000000u; // ga[56:32] | type=2
  i32x8 g1;
  g1[0] = 1 << 16;                  // workgroup_mask=0, data_size=1 (2 bytes)
  g1[1] = (int)(1024u << 16);       // tensor_dim0 = 1024 (bits 79:48, low half)
  g1[2] = (int)(16384u << 16);      // tensor_dim0 hi=0 | tensor_dim1 lo = 16384
  g1[3] = (int)(32u << 16);         // tensor_dim1 hi=0 | tile_dim0 = 32
  g1[4] = 128;                      // tile_dim1 = 128, tile_dim2 = 0
  g1[5] = 1024;                     // tensor_dim0_stride = 1024 (low 32)
  g1[6] = 0;                        // stride0 hi | tensor_dim1_stride lo
  g1[7] = 0;
  i32x4 z4 = {0, 0, 0, 0};
#if __clang_major__ >= 23
  i32x8 z8 = {0, 0, 0, 0, 0, 0, 0, 0};
  __builtin_amdgcn_tensor_load_to_lds(g0, g1, z4, z4, z8, 0);
#else
  __builtin_amdgcn_tensor_load_to_lds(g0, g1, z4, z4, 0);
#endif
}

// ---------- kernel 1: H -> bf16 and transposed bf16 copy ----------

__global__ void k_convert_h(const float* __restrict__ h,
                            unsigned short* __restrict__ hbf,
                            unsigned short* __restrict__ hbt) {
  size_t i = (size_t)blockIdx.x * blockDim.x + threadIdx.x; // < B*S*D
  unsigned short v = f2bf(h[i]);
  hbf[i] = v;
  unsigned int b = (unsigned int)(i >> 21);       // S*D = 2^21
  unsigned int s = (unsigned int)((i >> 10) & 2047);
  unsigned int d = (unsigned int)(i & 1023);
  hbt[((size_t)b * DQ + d) * SQ + s] = v;
}

__global__ void k_convert_wdec(const float* __restrict__ w,
                               unsigned short* __restrict__ wt) {
  size_t i = (size_t)blockIdx.x * blockDim.x + threadIdx.x; // < D*D
  unsigned int k = (unsigned int)(i >> 10), n = (unsigned int)(i & 1023);
  wt[(size_t)n * DQ + k] = f2bf(w[i]); // transpose: Wt[n][k] = W[k][n]
}

__global__ void k_convert_wout(const float* __restrict__ w,
                               unsigned short* __restrict__ wo) {
  size_t i = (size_t)blockIdx.x * blockDim.x + threadIdx.x; // < D*2D
  wo[i] = f2bf(w[i]);
}

// ---------- kernel 3: Q = H @ Wdec  (bf16 in, bf16 out, f32 acc) ----------

__global__ void __launch_bounds__(256)
k_gemm_q(const unsigned short* __restrict__ hbf,
         const unsigned short* __restrict__ wdt,
         unsigned short* __restrict__ qbf) {
  const int lane = threadIdx.x & 31;
  const int w = threadIdx.x >> 5;
  const int mw = w & 3, nw = w >> 2;
  const int row0 = blockIdx.x * 128 + mw * 32;
  const int col0 = blockIdx.y * 128 + nw * 64;

  v8f zero = {};
  v8f acc[2][4];
#pragma unroll
  for (int ms = 0; ms < 2; ++ms)
#pragma unroll
    for (int ns = 0; ns < 4; ++ns) acc[ms][ns] = zero;

  for (int k0 = 0; k0 < DQ; k0 += 32) {
    // lookahead prefetch of next K tile (global_prefetch_b8)
    if (k0 + 32 < DQ) {
      __builtin_prefetch(hbf + (size_t)row0 * DQ + k0 + 32, 0, 1);
      __builtin_prefetch(wdt + (size_t)col0 * DQ + k0 + 32, 0, 1);
    }
    v16bf a0 = load_a_rowmajor(hbf + (size_t)row0 * DQ + k0, DQ, lane);
    v16bf a1 = load_a_rowmajor(hbf + (size_t)(row0 + 16) * DQ + k0, DQ, lane);
#pragma unroll
    for (int ns = 0; ns < 4; ++ns) {
      v16bf bb = load_b_kmajor(wdt + (size_t)(col0 + ns * 16) * DQ + k0, DQ, lane);
      acc[0][ns] = wmma_bf16(a0, bb, acc[0][ns]);
      acc[1][ns] = wmma_bf16(a1, bb, acc[1][ns]);
    }
  }
  const int n = lane & 15, hi = lane >> 4;
#pragma unroll
  for (int ms = 0; ms < 2; ++ms)
#pragma unroll
    for (int ns = 0; ns < 4; ++ns)
#pragma unroll
      for (int v = 0; v < 8; ++v) {
        int row = row0 + ms * 16 + v + 8 * hi;
        int col = col0 + ns * 16 + n;
        qbf[(size_t)row * DQ + col] = f2bf(acc[ms][ns][v]);
      }
}

// ---------- kernel 4: streaming causal attention ----------
// block = 8 waves; 16 query rows per block; wave w owns d-slice [w*128, w*128+128)
// and K-slice [w*128, w*128+128) of the QK^T reduction.

__global__ void __launch_bounds__(256)
k_attn(const unsigned short* __restrict__ qbf,
       const unsigned short* __restrict__ hbf,
       const unsigned short* __restrict__ hbt,
       unsigned short* __restrict__ cbf) {
  __shared__ float sS[16][32];
  __shared__ unsigned short sP[16][32];
  __shared__ float sM[16], sL[16], sC[16];

  const int lane = threadIdx.x & 31;
  const int w = threadIdx.x >> 5;
  const int i0 = blockIdx.x * 16;
  const int b = blockIdx.y;
  const int n15 = lane & 15, hi = lane >> 4;

  if (threadIdx.x < 16) { sM[threadIdx.x] = -1e30f; sL[threadIdx.x] = 0.f; }

  // this wave's Q fragments (16 rows x 128 K-slice) — resident in VGPRs
  const int kw = w * 128;
  const unsigned short* qbase = qbf + ((size_t)b * SQ + i0) * DQ + kw;
  v16bf qa[4];
#pragma unroll
  for (int kc = 0; kc < 4; ++kc)
    qa[kc] = load_a_rowmajor(qbase + kc * 32, DQ, lane);

  v8f zero = {};
  v8f acc[8];
#pragma unroll
  for (int t = 0; t < 8; ++t) acc[t] = zero;

  __syncthreads();

  for (int j0 = 0; j0 <= i0; j0 += 32) {
    // zero score tile
    ((float*)sS)[threadIdx.x] = 0.f;
    ((float*)sS)[threadIdx.x + 256] = 0.f;
    __syncthreads();

    // partial S[16x32] over this wave's 128-wide K slice
#pragma unroll
    for (int js = 0; js < 2; ++js) {
      v8f p = zero;
      const unsigned short* hb =
          hbf + ((size_t)b * SQ + (j0 + js * 16)) * DQ + kw;
#pragma unroll
      for (int kc = 0; kc < 4; ++kc) {
        v16bf bb = load_b_kmajor(hb + kc * 32, DQ, lane); // B[k][j]=H[j][k]
        p = wmma_bf16(qa[kc], bb, p);
      }
#pragma unroll
      for (int v = 0; v < 8; ++v)
        atomicAdd(&sS[v + 8 * hi][js * 16 + n15], p[v]); // ds_add_f32
    }
    __syncthreads();

    // streaming softmax for 16 rows (wave 0, lanes 0..15)
    if (threadIdx.x < 16) {
      const int r = threadIdx.x;
      const int ig = i0 + r;
      float mold = sM[r];
      float mnew = mold;
      float sv[32];
#pragma unroll
      for (int c = 0; c < 32; ++c) {
        int j = j0 + c;
        bool ok = (j < ig) || (ig == 0 && j == 0); // strict causal + (0,0)
        float s = ok ? sS[r][c] : -1e30f;
        sv[c] = s;
        mnew = fmaxf(mnew, s);
      }
      float cf = __expf(mold - mnew);
      float ls = 0.f;
#pragma unroll
      for (int c = 0; c < 32; ++c) {
        float pe = __expf(sv[c] - mnew); // masked -> underflow to 0
        sP[r][c] = f2bf(pe);
        ls += pe;
      }
      sM[r] = mnew;
      sL[r] = sL[r] * cf + ls;
      sC[r] = cf;
    }
    __syncthreads();

    // rescale running context accumulator
    float cr[8];
#pragma unroll
    for (int v = 0; v < 8; ++v) cr[v] = sC[v + 8 * hi];
#pragma unroll
    for (int t = 0; t < 8; ++t)
#pragma unroll
      for (int v = 0; v < 8; ++v) acc[t][v] *= cr[v];

    // context += P(16x32) @ H(32 x 128-slice), B from transposed H (contig K)
    v16bf pa = load_a_rowmajor(&sP[0][0], 32, lane); // ds_load
    const unsigned short* hbt_b = hbt + (size_t)b * DQ * SQ;
#pragma unroll
    for (int t = 0; t < 8; ++t) {
      const int d0 = w * 128 + t * 16;
      v16bf bb = load_b_kmajor(hbt_b + (size_t)d0 * SQ + j0, SQ, lane);
      acc[t] = wmma_bf16(pa, bb, acc[t]);
    }
    __syncthreads();
  }

  // finalize: divide by l, emit bf16 context
  float li[8];
#pragma unroll
  for (int v = 0; v < 8; ++v) li[v] = 1.0f / sL[v + 8 * hi];
  unsigned short* cout = cbf + ((size_t)b * SQ + i0) * DQ;
#pragma unroll
  for (int t = 0; t < 8; ++t) {
    int d0 = w * 128 + t * 16 + n15;
#pragma unroll
    for (int v = 0; v < 8; ++v)
      cout[(size_t)(v + 8 * hi) * DQ + d0] = f2bf(acc[t][v] * li[v]);
  }
}

// ---------- kernel 5: out = tanh([H,C] @ Wout^T + b) ----------
// A tiles (128 rows x 32 K of H then C) staged through LDS by the Tensor Data
// Mover, double-buffered; wave 0 issues TENSOR_LOAD_TO_LDS and synchronizes
// with s_wait_tensorcnt, all waves consume via ds reads after the barrier.

__global__ void __launch_bounds__(256)
k_gemm_out(const unsigned short* __restrict__ hbf,
           const unsigned short* __restrict__ cbf,
           const unsigned short* __restrict__ wob,
           const float* __restrict__ bout,
           float* __restrict__ out) {
  __shared__ unsigned short sA[2][128 * 32]; // 2 x 8KB A-tile buffers

  const int lane = threadIdx.x & 31;
  const int w = threadIdx.x >> 5;
  const int mw = w & 3, nw = w >> 2;
  const int row0 = blockIdx.x * 128;   // block's 128 rows
  const int rw0 = mw * 32;             // wave's rows within the tile
  const int col0 = blockIdx.y * 128 + nw * 64;
  const int NSTEP = (2 * DQ) / 32;     // 64 K-steps (H for t<32, C after)

  v8f zero = {};
  v8f acc[2][4];
#pragma unroll
  for (int ms = 0; ms < 2; ++ms)
#pragma unroll
    for (int ns = 0; ns < 4; ++ns) acc[ms][ns] = zero;

  // prologue: DMA tile 0
  if (w == 0)
    tdm_load_tile_a(sA[0], hbf + (size_t)row0 * DQ);

  for (int t = 0; t < NSTEP; ++t) {
    if (w == 0) {
      if (t + 1 < NSTEP) {
        const int tn = t + 1;
        const unsigned short* src = (tn < 32)
            ? (hbf + (size_t)row0 * DQ + tn * 32)
            : (cbf + (size_t)row0 * DQ + (tn - 32) * 32);
        tdm_load_tile_a(sA[tn & 1], src);           // overlap DMA with math
        __builtin_amdgcn_s_wait_tensorcnt((short)1); // tile t complete
      } else {
        __builtin_amdgcn_s_wait_tensorcnt((short)0);
      }
    }
    __syncthreads();

    const unsigned short* at = sA[t & 1];
    const int k0 = t * 32;
    if (t + 1 < NSTEP) // keep B stream ahead in cache
      __builtin_prefetch(wob + (size_t)col0 * (2 * DQ) + k0 + 32, 0, 1);

    v16bf a0 = load_a_rowmajor(at + (size_t)rw0 * 32, 32, lane);       // ds
    v16bf a1 = load_a_rowmajor(at + (size_t)(rw0 + 16) * 32, 32, lane);
#pragma unroll
    for (int ns = 0; ns < 4; ++ns) {
      // B[k][n] = w_out[n][k] : row-major w_out is already contiguous in k
      v16bf bb = load_b_kmajor(wob + (size_t)(col0 + ns * 16) * (2 * DQ) + k0,
                               2 * DQ, lane);
      acc[0][ns] = wmma_bf16(a0, bb, acc[0][ns]);
      acc[1][ns] = wmma_bf16(a1, bb, acc[1][ns]);
    }
    __syncthreads(); // buffer t&1 free for reuse at t+2
  }

  const int n = lane & 15, hi = lane >> 4;
#pragma unroll
  for (int ms = 0; ms < 2; ++ms)
#pragma unroll
    for (int ns = 0; ns < 4; ++ns)
#pragma unroll
      for (int v = 0; v < 8; ++v) {
        int row = row0 + rw0 + ms * 16 + v + 8 * hi;
        int col = col0 + ns * 16 + n;
        out[(size_t)row * DQ + col] = tanhf(acc[ms][ns][v] + bout[col]);
      }
}

// ---------- launch ----------

extern "C" void kernel_launch(void* const* d_in, const int* in_sizes, int n_in,
                              void* d_out, int out_size, void* d_ws,
                              size_t ws_size, hipStream_t stream) {
  const float* h    = (const float*)d_in[0]; // [8,2048,1024]
  const float* wdec = (const float*)d_in[1]; // [1024,1024]
  const float* wout = (const float*)d_in[2]; // [1024,2048]
  const float* bout = (const float*)d_in[3]; // [1024]
  float* out = (float*)d_out;

  char* ws = (char*)d_ws;
  const size_t MB = 1ull << 20;
  unsigned short* hbf = (unsigned short*)(ws);             // 32 MB
  unsigned short* hbt = (unsigned short*)(ws + 32 * MB);   // 32 MB (transposed)
  unsigned short* qbf = (unsigned short*)(ws + 64 * MB);   // 32 MB
  unsigned short* cbf = (unsigned short*)(ws + 96 * MB);   // 32 MB
  unsigned short* wdt = (unsigned short*)(ws + 128 * MB);  // 2 MB
  unsigned short* wob = (unsigned short*)(ws + 130 * MB);  // 4 MB

  k_convert_h<<<(MTOT * DQ) / 256, 256, 0, stream>>>(h, hbf, hbt);
  k_convert_wdec<<<(DQ * DQ) / 256, 256, 0, stream>>>(wdec, wdt);
  k_convert_wout<<<(DQ * 2 * DQ) / 256, 256, 0, stream>>>(wout, wob);

  k_gemm_q<<<dim3(MTOT / 128, DQ / 128), 256, 0, stream>>>(hbf, wdt, qbf);
  k_attn<<<dim3(SQ / 16, BQ), 256, 0, stream>>>(qbf, hbf, hbt, cbf);
  k_gemm_out<<<dim3(MTOT / 128, DQ / 128), 256, 0, stream>>>(hbf, cbf, wob,
                                                             bout, out);
}
